// Generator_causal_54846732370488
// MI455X (gfx1250) — matrix operand order repaired and checked by reading, not compiled, not run on hardware
//
#include <hip/hip_runtime.h>

// CDNA5 / gfx1250, wave32. D = A(16x32 f16) x B(32x16 f16) + C(16x16 f32)
typedef __attribute__((ext_vector_type(16))) _Float16 v16h;
typedef __attribute__((ext_vector_type(8)))  float    v8f;

#ifndef __has_builtin
#define __has_builtin(x) 0
#endif
#if __has_builtin(__builtin_amdgcn_global_load_async_to_lds_b128)
#define USE_ASYNC_LDS 1
#else
#define USE_ASYNC_LDS 0
#endif

// Types matching the builtin prototype leaked by the round-2 diagnostic:
// param is 'int __vector(4) __device__ *', i.e. vector_size(16) int in AS(1).
typedef int v4i_ __attribute__((vector_size(16)));
typedef __attribute__((address_space(1))) v4i_* gp128_t;  // global src
typedef __attribute__((address_space(3))) v4i_* lp128_t;  // LDS dst

union AF { v16h v; uint4 q[2]; };

// A-matrix 16x32 f16 fragment (row-major source, lda in halves).
// ISA layout: lanes 0-15: M=lane, elems[0..7]=K0..7, [8..15]=K16..23
//             lanes16-31: M=lane-16, elems[0..7]=K8..15, [8..15]=K24..31
__device__ inline v16h frag_a(const _Float16* base, int lda, int lane) {
  int m    = lane & 15;
  int koff = (lane < 16) ? 0 : 8;
  const _Float16* p = base + m * lda + koff;
  AF f;
  f.q[0] = *(const uint4*)(p);
  f.q[1] = *(const uint4*)(p + 16);
  return f.v;
}

// B-matrix 32x16 f16 fragment. Weights stored as W[n][k] (k contiguous), ldw in halves.
// ISA layout: lanes 0-15: N=lane, elems[0..7]=K0..7, [8..15]=K16..23; lanes 16-31: N=lane-16, elems[0..7]=K8..15, [8..15]=K24..31
__device__ inline v16h frag_b(const _Float16* base, int ldw, int lane) {
  int n    = lane & 15;
  int koff = (lane < 16) ? 0 : 8;
  const _Float16* p = base + n * ldw + koff;
  AF f;
  f.q[0] = *(const uint4*)(p);
  f.q[1] = *(const uint4*)(p + 16);
  return f.v;
}

// One 128->128 layer: dst = relu(src @ W^T + bias). src/dst: [64][128] f16 LDS,
// W: [128][128] f16 LDS (n-major), bias: [128] f32 global (L2-hot).
__device__ inline void mlp128(const _Float16* __restrict__ src,
                              const _Float16* __restrict__ W,
                              const float* __restrict__ bias,
                              _Float16* __restrict__ dst,
                              int wave, int lane) {
  const int mt   = wave >> 1;                       // 4 M-tiles over 8 waves
  const int npar = wave & 1;
  const int nlo  = lane & 15;
  const int moff = mt * 16 + ((lane < 16) ? 0 : 8); // C layout: VGPR v -> row moff+v
  v8f acc[4];
  for (int nn = 0; nn < 4; ++nn) {
    float bv = bias[(npar + nn * 2) * 16 + nlo];
    v8f c;
    for (int v = 0; v < 8; ++v) c[v] = bv;
    acc[nn] = c;
  }
  for (int kc = 0; kc < 4; ++kc) {                  // K = 128 in 4 chunks of 32
    v16h a = frag_a(src + mt * 16 * 128 + kc * 32, 128, lane);
    for (int nn = 0; nn < 4; ++nn) {
      int nt = npar + nn * 2;
      v16h b = frag_b(W + nt * 16 * 128 + kc * 32, 128, lane);
      acc[nn] = __builtin_amdgcn_wmma_f32_16x16x32_f16(
          false, a, false, b, (short)0, acc[nn], false, false);
    }
  }
  for (int nn = 0; nn < 4; ++nn) {
    int n = (npar + nn * 2) * 16 + nlo;
    for (int v = 0; v < 8; ++v)
      dst[(moff + v) * 128 + n] = (_Float16)fmaxf(acc[nn][v], 0.f);
  }
}

// Pre-pass: f32 -> f16 weight conversion into workspace.
// Wi [64][128][65] -> Wi_h [64][128][64] (k=0..63) + wz [64][128] (the z column, f32)
__global__ __launch_bounds__(256) void convert_weights_kernel(
    const float* __restrict__ Wi, const float* __restrict__ Ws1,
    const float* __restrict__ Ws2, _Float16* __restrict__ Wi_h,
    _Float16* __restrict__ Ws1_h, _Float16* __restrict__ Ws2_h,
    float* __restrict__ wz) {
  int t = blockIdx.x * blockDim.x + threadIdx.x;
  if (t < 64 * 128 * 64) {
    int k = t & 63, n = (t >> 6) & 127, i = t >> 13;
    Wi_h[t] = (_Float16)Wi[i * 8320 + n * 65 + k];
  }
  if (t < 128 * 128) {
    Ws1_h[t] = (_Float16)Ws1[t];
    Ws2_h[t] = (_Float16)Ws2[t];
  }
  if (t < 64 * 128) {
    int n = t & 127, i = t >> 7;
    wz[t] = Wi[i * 8320 + n * 65 + 64];
  }
}

// Persistent per-block causal scan: each block owns 64 batch rows; the 64-node
// recurrence runs entirely in LDS (rows are independent across blocks).
// Per-step input weights are double-buffered and prefetched asynchronously
// (global->LDS DMA, ASYNCcnt) so L2 latency overlaps the WMMA layers.
__global__ __launch_bounds__(256) void gen_causal_kernel(
    const float* __restrict__ x, const float* __restrict__ z,
    const float* __restrict__ M, const float* __restrict__ bi,
    const float* __restrict__ bs1, const float* __restrict__ bs2,
    const float* __restrict__ wf, const float* __restrict__ bf,
    const _Float16* __restrict__ Wi_h, const _Float16* __restrict__ Ws1_h,
    const _Float16* __restrict__ Ws2_h, const float* __restrict__ wz,
    float* __restrict__ out) {
  __shared__ _Float16 Ws1_s[128 * 128];     // 32 KB
  __shared__ _Float16 Ws2_s[128 * 128];     // 32 KB
  __shared__ _Float16 WiBuf[2 * 128 * 64];  // 32 KB (double-buffered per step)
  __shared__ float    out_s[64 * 64];       // 16 KB (the evolving out tile)
  __shared__ _Float16 xm_s[64 * 64];        //  8 KB
  __shared__ _Float16 h1_s[64 * 128];       // 16 KB (ping)
  __shared__ _Float16 h2_s[64 * 128];       // 16 KB (pong)
  __shared__ float    z_s[64];
  __shared__ float    red_s[64 * 4];        //  1 KB

  const int  tid  = threadIdx.x;
  const int  lane = tid & 31;
  const int  wave = tid >> 5;
  const long b0   = (long)blockIdx.x * 64;
  const int  jcol = tid & 63;               // column this thread owns in xm

  for (int t = tid; t < 64 * 64; t += 256) out_s[t] = x[b0 * 64 + t];
  {
    const uint4* g1 = (const uint4*)Ws1_h; uint4* s1 = (uint4*)Ws1_s;
    const uint4* g2 = (const uint4*)Ws2_h; uint4* s2 = (uint4*)Ws2_s;
    for (int t = tid; t < 128 * 128 / 8; t += 256) { s1[t] = g1[t]; s2[t] = g2[t]; }
    const uint4* g0 = (const uint4*)Wi_h;  uint4* s0 = (uint4*)WiBuf;
    for (int t = tid; t < 128 * 64 / 8; t += 256) s0[t] = g0[t]; // step 0 weights
  }
  __syncthreads();

  for (int i = 0; i < 64; ++i) {
    const int cur = i & 1, nxt = cur ^ 1;
    const int inext = (i < 63) ? i + 1 : 63;
    const _Float16* Wi_s = WiBuf + cur * 8192;

    // stage z column + masked parents. Each thread's xm column index is
    // invariant (stride 256 == 0 mod 64) -> exactly one M[j,i] load (L1-hot).
    if (tid < 64) z_s[tid] = z[(b0 + tid) * 64 + i];
    {
      float mj = M[jcol * 64 + i];          // xm = out * M[:, i]
      if (jcol == i) mj = 0.f;              // x_masked[:, i] = 0 (diag is 0 anyway)
      for (int t = tid; t < 64 * 64; t += 256)
        xm_s[t] = (_Float16)(out_s[t] * mj);
    }
    __syncthreads();

    // ---- prefetch next step's input weights into the back buffer ----
#if USE_ASYNC_LDS
    {
      const uint4* g = (const uint4*)(Wi_h + (long)inext * 8192);
      uint4*       s = (uint4*)(WiBuf + nxt * 8192);
      for (int q = 0; q < 4; ++q) {
        int t = tid + q * 256;
        __builtin_amdgcn_global_load_async_to_lds_b128(
            (gp128_t)(const void*)(g + t), (lp128_t)(void*)(s + t), 0, 0);
      }
    }
#else
    uint4 pre[4];
    {
      const uint4* g = (const uint4*)(Wi_h + (long)inext * 8192);
      for (int q = 0; q < 4; ++q) pre[q] = g[tid + q * 256];
    }
#endif

    { // layer 1: h1 = relu(xm @ Wi_i[:, :64]^T + z*wz_i + bi_i), K=64
      const int mt = wave >> 1, npar = wave & 1, nlo = lane & 15;
      const int moff = mt * 16 + ((lane < 16) ? 0 : 8);
      v8f acc[4];
      for (int nn = 0; nn < 4; ++nn) {
        int   n   = (npar + nn * 2) * 16 + nlo;
        float wzv = wz[i * 128 + n];
        float bv  = bi[i * 128 + n];
        v8f c;
        for (int v = 0; v < 8; ++v) c[v] = z_s[moff + v] * wzv + bv; // rank-1 z term
        acc[nn] = c;
      }
      for (int kc = 0; kc < 2; ++kc) {
        v16h a = frag_a(xm_s + mt * 16 * 64 + kc * 32, 64, lane);
        for (int nn = 0; nn < 4; ++nn) {
          int nt = npar + nn * 2;
          v16h b = frag_b(Wi_s + nt * 16 * 64 + kc * 32, 64, lane);
          acc[nn] = __builtin_amdgcn_wmma_f32_16x16x32_f16(
              false, a, false, b, (short)0, acc[nn], false, false);
        }
      }
      for (int nn = 0; nn < 4; ++nn) {
        int n = (npar + nn * 2) * 16 + nlo;
        for (int v = 0; v < 8; ++v)
          h1_s[(moff + v) * 128 + n] = (_Float16)fmaxf(acc[nn][v], 0.f);
      }
    }
    __syncthreads();
    mlp128(h1_s, Ws1_s, bs1, h2_s, wave, lane);   // shared block 1
    __syncthreads();
    mlp128(h2_s, Ws2_s, bs2, h1_s, wave, lane);   // shared block 2 (h3 -> h1 buffer)
    __syncthreads();

    { // head: out[:, i] = sigmoid(h3 @ wf_i + bf_i)
      int r = tid >> 2, p = tid & 3;
      const _Float16* hrow = h1_s + r * 128 + p * 32;
      const float*    wrow = wf + i * 128 + p * 32;
      float s = 0.f;
      for (int k = 0; k < 32; ++k) s += (float)hrow[k] * wrow[k];
      red_s[r * 4 + p] = s;
    }
    __syncthreads();
    if (tid < 64) {
      float s = red_s[tid * 4] + red_s[tid * 4 + 1] + red_s[tid * 4 + 2] +
                red_s[tid * 4 + 3] + bf[i];
      out_s[tid * 64 + i] = 1.f / (1.f + __expf(-s));
    }

    // drain the weight prefetch before the end-of-step barrier publishes it
#if USE_ASYNC_LDS
    asm volatile("s_wait_asynccnt 0" ::: "memory");
#else
    {
      uint4* s = (uint4*)(WiBuf + nxt * 8192);
      for (int q = 0; q < 4; ++q) s[tid + q * 256] = pre[q];
    }
#endif
    __syncthreads();
  }

  for (int t = tid; t < 64 * 64; t += 256) out[b0 * 64 + t] = out_s[t];
}

extern "C" void kernel_launch(void* const* d_in, const int* in_sizes, int n_in,
                              void* d_out, int out_size, void* d_ws, size_t ws_size,
                              hipStream_t stream) {
  const float* x   = (const float*)d_in[0];
  const float* z   = (const float*)d_in[1];
  const float* M   = (const float*)d_in[2];
  const float* Wi  = (const float*)d_in[3];
  const float* bi  = (const float*)d_in[4];
  const float* Ws1 = (const float*)d_in[5];
  const float* bs1 = (const float*)d_in[6];
  const float* Ws2 = (const float*)d_in[7];
  const float* bs2 = (const float*)d_in[8];
  const float* wf  = (const float*)d_in[9];
  const float* bf  = (const float*)d_in[10];
  float* out = (float*)d_out;

  char* ws = (char*)d_ws;
  _Float16* Wi_h  = (_Float16*)(ws);                      // 64*128*64*2 = 1 MiB
  _Float16* Ws1_h = (_Float16*)(ws + 1048576);            // 32 KiB
  _Float16* Ws2_h = (_Float16*)(ws + 1048576 + 32768);    // 32 KiB
  float*    wz    = (float*)   (ws + 1048576 + 65536);    // 32 KiB

  convert_weights_kernel<<<2048, 256, 0, stream>>>(Wi, Ws1, Ws2, Wi_h, Ws1_h,
                                                   Ws2_h, wz);
  gen_causal_kernel<<<65536 / 64, 256, 0, stream>>>(x, z, M, bi, bs1, bs2, wf,
                                                    bf, Wi_h, Ws1_h, Ws2_h, wz,
                                                    out);
}
